// MemTransformerLM_55439437857183
// MI455X (gfx1250) — matrix operand order repaired and checked by reading, not compiled, not run on hardware
//
#include <hip/hip_runtime.h>

typedef __bf16 bf16;
typedef __bf16 v16bf  __attribute__((ext_vector_type(16)));
typedef __bf16 bf16x8 __attribute__((ext_vector_type(8)));
typedef float  v8f    __attribute__((ext_vector_type(8)));

#define QLEN 1024
#define MLEN 1024
#define KLEN 2048
#define BSZ 4
#define DM 1024
#define NH 16
#define DH 64

// ---------------------------------------------------------------------------
// WMMA helpers (CDNA5 gfx1250, wave32). Layouts per CDNA5 ISA:
//  A 16x32 bf16: lanes 0-15 row M=lane, K {0..7,16..23}; lanes 16-31 row
//    M=lane-16, K {8..15,24..31}  -> two 16B loads at k0+hi*8 and +16.
//  B 32x16 bf16 (from B^T row-major [N,K]): lane col N=lane&15, same K as A:
//    {0..7,16..23} (hi=0) / {8..15,24..31} (hi=1) -> two 16B loads.
//  C/D f32 16x16: VGPR v holds row v+8*hi, col lane&15.
// ---------------------------------------------------------------------------
__device__ __forceinline__ v16bf load_a16x32(const bf16* __restrict__ base,
                                             int ld, int row0, int k0) {
  const int lane = threadIdx.x & 31;
  const int lo = lane & 15, hi = lane >> 4;
  const bf16* p = base + (size_t)(row0 + lo) * ld + k0 + hi * 8;
  union { v16bf v; bf16x8 h[2]; } u;
  u.h[0] = *(const bf16x8*)(p);
  u.h[1] = *(const bf16x8*)(p + 16);
  return u.v;
}

__device__ __forceinline__ v16bf load_b32x16(const bf16* __restrict__ baseT,
                                             int ld, int k0, int n0) {
  const int lane = threadIdx.x & 31;
  const int lo = lane & 15, hi = lane >> 4;
  const bf16* p = baseT + (size_t)(n0 + lo) * ld + k0 + hi * 8;
  union { v16bf v; bf16x8 h[2]; } u;
  u.h[0] = *(const bf16x8*)(p);
  u.h[1] = *(const bf16x8*)(p + 16);
  return u.v;
}

__device__ __forceinline__ v8f wmma_bf16(v16bf a, v16bf b, v8f c) {
  return __builtin_amdgcn_wmma_f32_16x16x32_bf16(false, a, false, b,
                                                 (short)0, c, false, false);
}

// ---------------------------------------------------------------------------
// Elementwise conversions
// ---------------------------------------------------------------------------
__global__ void cvt_f32_bf16(const float* __restrict__ s, bf16* __restrict__ d,
                             int n) {
  int i = blockIdx.x * 256 + threadIdx.x;
  if (i < n) d[i] = (bf16)s[i];
}

// src [T][BSZ][DM] time-major -> dst batch-major rows: cat[b*KLEN + tOff + t][d]
__global__ void cvt_cat(const float* __restrict__ s, bf16* __restrict__ d,
                        int tOff, int n) {
  int i = blockIdx.x * 256 + threadIdx.x;
  if (i >= n) return;
  int t = i >> 12;          // 4096 elements per time step
  int rem = i & 4095;
  int b = rem >> 10;
  int dd = rem & 1023;
  d[((size_t)(b * KLEN + tOff + t)) * DM + dd] = (bf16)s[i];
}

// dst[n][k] = src[k][n], 1024x1024, bf16 out
__global__ void transpose_cvt_1024(const float* __restrict__ s,
                                   bf16* __restrict__ d) {
  int n = blockIdx.x * 16 + threadIdx.x;
  int k = blockIdx.y * 16 + threadIdx.y;
  d[(size_t)n * 1024 + k] = (bf16)s[(size_t)k * 1024 + n];
}

// ---------------------------------------------------------------------------
// Generic K=1024 bf16 GEMM: C[M,1024] = A[M,1024] @ Bt^T  (Bt stored [N,K]).
// Block = 8 waves, each wave computes a 16x64 strip.
// Row semantics: MODE 0/1 -> A=cat (batch-major): m = b*KLEN + j
//                MODE 2   -> A=cat, GEMM row m (b*QLEN+i) remapped to
//                            cat row b*KLEN + MLEN + i
//                MODE 3   -> m = c (r rows);  MODE 4 -> m = i*BSZ+b (t-major)
// Epilogues:
//   0: K  -> k_ws  [b][h][j][d]   bf16, +bias
//   1: V  -> vT_ws [b][h][d][j]   bf16, +bias, packed b128 stores
//   2: Q  -> qw/qr [b][h][i][d]   bf16, +bias + r_w_bias / r_r_bias
//   3: RK -> rk_ws [h][c][d]      bf16, +bias
//   4: O  -> f32 [m][1024] = acc + bias + resid[m][n]
// ---------------------------------------------------------------------------
template <int MODE>
__global__ __launch_bounds__(256) void gemm_bf16_k1024(
    const bf16* __restrict__ A, const bf16* __restrict__ Bt, int M,
    const float* __restrict__ bias, const float* __restrict__ bias2a,
    const float* __restrict__ bias2b, bf16* __restrict__ dst,
    bf16* __restrict__ dst2, float* __restrict__ dstf,
    const float* __restrict__ resid) {
  const int lane = threadIdx.x & 31;
  const int wid = threadIdx.x >> 5;
  const int lo = lane & 15, hi = lane >> 4;
  const int m0 = blockIdx.y * 128 + wid * 16;
  const int n0 = blockIdx.x * 64;
  if (m0 >= M) return;

  // A-row remap for Q (skip memory rows of batch-major cat)
  const int arow0 = (MODE == 2) ? ((m0 >> 10) * KLEN + MLEN + (m0 & 1023)) : m0;

  v8f acc[4] = {};
  for (int k0 = 0; k0 < 1024; k0 += 32) {
    v16bf a = load_a16x32(A, 1024, arow0, k0);
#pragma unroll
    for (int c = 0; c < 4; ++c) {
      v16bf b = load_b32x16(Bt, 1024, k0, n0 + c * 16);
      acc[c] = wmma_bf16(a, b, acc[c]);
    }
    if (k0 + 64 < 1024) {  // global_prefetch_b8 of the next-but-one panels
      __builtin_prefetch((const void*)(A + (size_t)(arow0 + lo) * 1024 + k0 + 64), 0, 0);
      __builtin_prefetch((const void*)(Bt + (size_t)(n0 + lo) * 1024 + k0 + 64), 0, 0);
    }
  }

  if (MODE == 1) {
    // packed epilogue: a lane's 8 accumulator rows are consecutive j
    const int mb = m0 + 8 * hi;        // first row this lane holds
    const int b = mb >> 11, j = mb & (KLEN - 1);
#pragma unroll
    for (int c = 0; c < 4; ++c) {
      const int n = n0 + c * 16 + lo;
      const int hh = n >> 6, d = n & 63;
      const float bz = bias[n];
      bf16x8 pk;
#pragma unroll
      for (int v = 0; v < 8; ++v) pk[v] = (bf16)(acc[c][v] + bz);
      *(bf16x8*)(dst + (((size_t)(b * NH + hh)) * DH + d) * KLEN + j) = pk;
    }
    return;
  }

#pragma unroll
  for (int c = 0; c < 4; ++c) {
#pragma unroll
    for (int v = 0; v < 8; ++v) {
      int m = m0 + v + 8 * hi;
      int n = n0 + c * 16 + lo;
      float val = acc[c][v] + bias[n];
      if (MODE == 0) {
        int b = m >> 11, j = m & (KLEN - 1), hh = n >> 6, d = n & 63;
        dst[(((size_t)(b * NH + hh)) * KLEN + j) * DH + d] = (bf16)val;
      } else if (MODE == 2) {
        int b = m >> 10, i = m & (QLEN - 1), hh = n >> 6, d = n & 63;
        size_t idx = (((size_t)(b * NH + hh)) * QLEN + i) * DH + d;
        dst[idx] = (bf16)(val + bias2a[n]);
        dst2[idx] = (bf16)(val + bias2b[n]);
      } else if (MODE == 3) {
        int hh = n >> 6, d = n & 63;
        dst[(((size_t)hh) * KLEN + m) * DH + d] = (bf16)val;
      } else {
        size_t idx = (size_t)m * 1024 + n;
        dstf[idx] = val + resid[idx];
      }
    }
  }
}

// ---------------------------------------------------------------------------
// BDrawT[b,h,c,i] = (q + r_r_bias)[b,h,i,:] . rk[h,c,:]   (K = 64)
// Output TRANSPOSED [bn][c][i] so a lane's 8 rows pack into one b128 store.
// ---------------------------------------------------------------------------
__global__ __launch_bounds__(256) void bd_gemm(const bf16* __restrict__ qr,
                                               const bf16* __restrict__ rk,
                                               bf16* __restrict__ bd) {
  const int lane = threadIdx.x & 31;
  const int wid = threadIdx.x >> 5;
  const int lo = lane & 15, hi = lane >> 4;
  const int bn = blockIdx.z;  // b*16+h
  const int h = bn & (NH - 1);
  const int m0 = blockIdx.y * 128 + wid * 16;
  const int n0 = blockIdx.x * 64;
  const bf16* Ah = qr + (size_t)bn * QLEN * DH;
  const bf16* Bh = rk + (size_t)h * KLEN * DH;
  bf16* out = bd + (size_t)bn * QLEN * KLEN;

  v16bf a0 = load_a16x32(Ah, DH, m0, 0);
  v16bf a1 = load_a16x32(Ah, DH, m0, 32);
  const int mb = m0 + 8 * hi;
#pragma unroll
  for (int c = 0; c < 4; ++c) {
    v16bf b0 = load_b32x16(Bh, DH, 0, n0 + c * 16);
    v16bf b1 = load_b32x16(Bh, DH, 32, n0 + c * 16);
    v8f acc = {};
    acc = wmma_bf16(a0, b0, acc);
    acc = wmma_bf16(a1, b1, acc);
    const int n = n0 + c * 16 + lo;
    bf16x8 pk;
#pragma unroll
    for (int v = 0; v < 8; ++v) pk[v] = (bf16)acc[v];
    *(bf16x8*)(out + (size_t)n * QLEN + mb) = pk;  // [c][i]
  }
}

// ---------------------------------------------------------------------------
// Flash attention: one wave per 16-row q-tile per (b,h).
// score = ((qw . k^T) + BDrawT[j+QLEN-1-i, i]) * scale, masked j <= i+MLEN,
// online softmax, O += P @ V (V pre-transposed).
// ---------------------------------------------------------------------------
#define FA_WAVES 4
__global__ __launch_bounds__(32 * FA_WAVES) void flash_attn(
    const bf16* __restrict__ qw, const bf16* __restrict__ kh,
    const bf16* __restrict__ vT, const bf16* __restrict__ bd,
    bf16* __restrict__ attn_vec) {
  __shared__ __align__(16) bf16 pbuf[FA_WAVES][16 * 32];
  const int lane = threadIdx.x & 31;
  const int wid = threadIdx.x >> 5;
  const int lo = lane & 15, hi = lane >> 4;
  const int b = blockIdx.y, h = blockIdx.z;
  const int bn = b * NH + h;
  const int i0 = (blockIdx.x * FA_WAVES + wid) * 16;

  const bf16* qh = qw + (size_t)bn * QLEN * DH;
  const bf16* kp = kh + (size_t)bn * KLEN * DH;
  const bf16* vp = vT + (size_t)bn * DH * KLEN;
  const bf16* bdp = bd + (size_t)bn * QLEN * KLEN;  // [c][i]

  v16bf aq0 = load_a16x32(qh, DH, i0, 0);
  v16bf aq1 = load_a16x32(qh, DH, i0, 32);

  float rmax[8], rsum[8];
  v8f acc[4] = {};
#pragma unroll
  for (int v = 0; v < 8; ++v) {
    rmax[v] = -__builtin_inff();
    rsum[v] = 0.f;
  }

  const float scale = 0.125f;  // 1/sqrt(64)
  const int jmax = i0 + 15 + MLEN;
  for (int j0 = 0; j0 <= jmax; j0 += 32) {
    v8f s[2];
#pragma unroll
    for (int t = 0; t < 2; ++t) {
      v16bf bk0 = load_b32x16(kp, DH, 0, j0 + t * 16);
      v16bf bk1 = load_b32x16(kp, DH, 32, j0 + t * 16);
      v8f cs = {};
      cs = wmma_bf16(aq0, bk0, cs);
      cs = wmma_bf16(aq1, bk1, cs);
      s[t] = cs;
    }
#pragma unroll
    for (int v = 0; v < 8; ++v) {
      int rowg = i0 + v + 8 * hi;
      float sv[2];
#pragma unroll
      for (int t = 0; t < 2; ++t) {
        int col = j0 + t * 16 + lo;
        if (col <= rowg + MLEN) {            // causal-with-memory mask
          int cidx = col + (QLEN - 1) - rowg;  // rel-shift gather
          float bdv = (float)bdp[(size_t)cidx * QLEN + rowg];
          sv[t] = (s[t][v] + bdv) * scale;
        } else {
          sv[t] = -__builtin_inff();
        }
      }
      float mx = fmaxf(sv[0], sv[1]);
#pragma unroll
      for (int o = 1; o < 16; o <<= 1) mx = fmaxf(mx, __shfl_xor(mx, o, 32));
      mx = fmaxf(mx, rmax[v]);
      float corr = __expf(rmax[v] - mx);
      float p0 = __expf(sv[0] - mx);
      float p1 = __expf(sv[1] - mx);
      float ps = p0 + p1;
#pragma unroll
      for (int o = 1; o < 16; o <<= 1) ps += __shfl_xor(ps, o, 32);
      rsum[v] = rsum[v] * corr + ps;
      rmax[v] = mx;
#pragma unroll
      for (int c = 0; c < 4; ++c) acc[c][v] = acc[c][v] * corr;
      int r = v + 8 * hi;
      pbuf[wid][r * 32 + lo] = (bf16)p0;
      pbuf[wid][r * 32 + 16 + lo] = (bf16)p1;
    }
    // cross-lane transpose of P through wave-private LDS tile
    asm volatile("s_wait_dscnt 0" ::: "memory");
    union { v16bf v; bf16x8 h2[2]; } up;
    const bf16* pb = &pbuf[wid][0] + lo * 32 + hi * 8;
    up.h2[0] = *(const bf16x8*)(pb);
    up.h2[1] = *(const bf16x8*)(pb + 16);
    v16bf pa = up.v;
#pragma unroll
    for (int c = 0; c < 4; ++c) {
      v16bf bv = load_b32x16(vp, KLEN, j0, c * 16);
      acc[c] = wmma_bf16(pa, bv, acc[c]);
    }
    asm volatile("" ::: "memory");
  }

#pragma unroll
  for (int v = 0; v < 8; ++v) {
    int rowg = i0 + v + 8 * hi;
    float rinv = 1.0f / rsum[v];
#pragma unroll
    for (int c = 0; c < 4; ++c) {
      int d = c * 16 + lo;
      attn_vec[((size_t)(rowg * BSZ + b)) * DM + h * DH + d] =
          (bf16)(acc[c][v] * rinv);
    }
  }
}

// ---------------------------------------------------------------------------
// Row LayerNorm over d=1024, input already = w + attn_out + bo (f32)
// ---------------------------------------------------------------------------
__global__ __launch_bounds__(256) void layernorm_rows(
    const float* __restrict__ x, const float* __restrict__ g,
    const float* __restrict__ be, float* __restrict__ out) {
  __shared__ float red[256];
  int row = blockIdx.x;
  const float* xr = x + (size_t)row * DM;
  float s = 0.f, s2 = 0.f;
  for (int d = threadIdx.x; d < DM; d += 256) {
    float v = xr[d];
    s += v;
    s2 += v * v;
  }
  red[threadIdx.x] = s;
  __syncthreads();
  for (int o = 128; o > 0; o >>= 1) {
    if (threadIdx.x < o) red[threadIdx.x] += red[threadIdx.x + o];
    __syncthreads();
  }
  float mean = red[0] * (1.0f / DM);
  __syncthreads();
  red[threadIdx.x] = s2;
  __syncthreads();
  for (int o = 128; o > 0; o >>= 1) {
    if (threadIdx.x < o) red[threadIdx.x] += red[threadIdx.x + o];
    __syncthreads();
  }
  float var = red[0] * (1.0f / DM) - mean * mean;
  float rstd = rsqrtf(var + 1e-5f);
  for (int d = threadIdx.x; d < DM; d += 256) {
    out[(size_t)row * DM + d] = (xr[d] - mean) * rstd * g[d] + be[d];
  }
}

// ---------------------------------------------------------------------------
// Host orchestration
// ---------------------------------------------------------------------------
extern "C" void kernel_launch(void* const* d_in, const int* in_sizes, int n_in,
                              void* d_out, int out_size, void* d_ws,
                              size_t ws_size, hipStream_t stream) {
  (void)in_sizes; (void)n_in; (void)out_size; (void)ws_size;
  const float* w = (const float*)d_in[0];
  const float* r = (const float*)d_in[1];
  const float* rwb = (const float*)d_in[2];
  const float* rrb = (const float*)d_in[3];
  const float* mems = (const float*)d_in[4];
  const float* Wq = (const float*)d_in[5];
  const float* bq = (const float*)d_in[6];
  const float* Wk = (const float*)d_in[7];
  const float* bk = (const float*)d_in[8];
  const float* Wv = (const float*)d_in[9];
  const float* bv = (const float*)d_in[10];
  const float* Wr = (const float*)d_in[11];
  const float* br = (const float*)d_in[12];
  const float* Wo = (const float*)d_in[13];
  const float* bo = (const float*)d_in[14];
  const float* lng = (const float*)d_in[15];
  const float* lnb = (const float*)d_in[16];
  float* out = (float*)d_out;

  char* wsb = (char*)d_ws;
  size_t off = 0;
  auto alloc = [&](size_t bytes) -> void* {
    void* p = wsb + off;
    off = (off + bytes + 255) & ~(size_t)255;
    return p;
  };
  bf16* cat   = (bf16*)alloc((size_t)KLEN * BSZ * DM * 2);    // 16 MB, batch-major
  bf16* rbf   = (bf16*)alloc((size_t)KLEN * DM * 2);          //  4 MB
  bf16* WqT   = (bf16*)alloc((size_t)DM * DM * 2);
  bf16* WkT   = (bf16*)alloc((size_t)DM * DM * 2);
  bf16* WvT   = (bf16*)alloc((size_t)DM * DM * 2);
  bf16* WrT   = (bf16*)alloc((size_t)DM * DM * 2);
  bf16* WoT   = (bf16*)alloc((size_t)DM * DM * 2);
  bf16* k_ws  = (bf16*)alloc((size_t)BSZ * NH * KLEN * DH * 2);  // 16 MB
  bf16* vT_ws = (bf16*)alloc((size_t)BSZ * NH * DH * KLEN * 2);  // 16 MB
  bf16* qw    = (bf16*)alloc((size_t)BSZ * NH * QLEN * DH * 2);  //  8 MB
  bf16* qr    = (bf16*)alloc((size_t)BSZ * NH * QLEN * DH * 2);  //  8 MB
  bf16* rk_ws = (bf16*)alloc((size_t)NH * KLEN * DH * 2);        //  4 MB
  bf16* bdraw = (bf16*)alloc((size_t)BSZ * NH * QLEN * KLEN * 2);// 256 MB [bn][c][i]
  bf16* avec  = (bf16*)alloc((size_t)QLEN * BSZ * DM * 2);       //  8 MB
  float* aout = (float*)alloc((size_t)QLEN * BSZ * DM * 4);      // 16 MB

  const int nCat = MLEN * BSZ * DM;  // 4,194,304 elements per half
  cvt_cat<<<(nCat + 255) / 256, 256, 0, stream>>>(mems, cat, 0, nCat);
  cvt_cat<<<(nCat + 255) / 256, 256, 0, stream>>>(w, cat, MLEN, nCat);
  const int nR = KLEN * DM;
  cvt_f32_bf16<<<(nR + 255) / 256, 256, 0, stream>>>(r, rbf, nR);

  dim3 tb(16, 16), tg(64, 64);
  transpose_cvt_1024<<<tg, tb, 0, stream>>>(Wq, WqT);
  transpose_cvt_1024<<<tg, tb, 0, stream>>>(Wk, WkT);
  transpose_cvt_1024<<<tg, tb, 0, stream>>>(Wv, WvT);
  transpose_cvt_1024<<<tg, tb, 0, stream>>>(Wr, WrT);
  transpose_cvt_1024<<<tg, tb, 0, stream>>>(Wo, WoT);

  // Projections (all WMMA)
  gemm_bf16_k1024<0><<<dim3(16, 64), 256, 0, stream>>>(
      cat, WkT, KLEN * BSZ, bk, nullptr, nullptr, k_ws, nullptr, nullptr,
      nullptr);
  gemm_bf16_k1024<1><<<dim3(16, 64), 256, 0, stream>>>(
      cat, WvT, KLEN * BSZ, bv, nullptr, nullptr, vT_ws, nullptr, nullptr,
      nullptr);
  gemm_bf16_k1024<2><<<dim3(16, 32), 256, 0, stream>>>(
      cat, WqT, QLEN * BSZ, bq, rwb, rrb, qw, qr, nullptr, nullptr);
  gemm_bf16_k1024<3><<<dim3(16, 16), 256, 0, stream>>>(
      rbf, WrT, KLEN, br, nullptr, nullptr, rk_ws, nullptr, nullptr, nullptr);

  // BD term (pre rel-shift), WMMA K=64, transposed packed output
  bd_gemm<<<dim3(KLEN / 64, QLEN / 128, BSZ * NH), 256, 0, stream>>>(qr, rk_ws,
                                                                     bdraw);

  // Fused masked softmax attention with shifted-BD gather, WMMA QK^T and P@V
  flash_attn<<<dim3(QLEN / (16 * FA_WAVES), BSZ, NH), 32 * FA_WAVES, 0,
               stream>>>(qw, k_ws, vT_ws, bdraw, avec);

  // Output projection + bias + residual
  gemm_bf16_k1024<4><<<dim3(16, 32), 256, 0, stream>>>(
      avec, WoT, QLEN * BSZ, bo, nullptr, nullptr, nullptr, nullptr, aout, w);

  // LayerNorm -> final f32 output [qlen, bsz, d]
  layernorm_rows<<<QLEN * BSZ, 256, 0, stream>>>(aout, lng, lnb, out);
}
